// SwinTransformerBlock_28398323761706
// MI455X (gfx1250) — compile-verified, matrix operand order checked by reading, not run on hardware
//
#include <hip/hip_runtime.h>
#include <hip/hip_bf16.h>

// ---------------------------------------------------------------------------
// Swin block on MI455X (gfx1250): all GEMMs via v_wmma_f32_16x16x32_f16,
// attention per (window, head) with GLOBAL_LOAD_ASYNC_TO_LDS_B128 staging,
// precomputed bias+mask table (no divergent gathers in the hot loop),
// softmax via wave32 16-lane shfl reductions.
// ---------------------------------------------------------------------------

typedef _Float16 v16h __attribute__((ext_vector_type(16)));
typedef _Float16 v8h  __attribute__((ext_vector_type(8)));
typedef float    v8f  __attribute__((ext_vector_type(8)));

union Frag { v16h v; v8h h[2]; };

#define BATCH   32
#define LTOK    3136      // 56*56
#define DIMC    256
#define NHEAD   8
#define HDIM    32
#define NTOK    49        // 7*7
#define NWIN    64        // windows per image (8*8)
#define BWIN    2048      // BATCH*NWIN
#define MROWS   100352    // BWIN*NTOK = BATCH*LTOK
#define QSCALE  0.1767766952966369f   // 1/sqrt(32)

// low 32 bits of a generic pointer to a __shared__ object = LDS byte offset
// (ISA 10.2 aperture rules)
__device__ __forceinline__ unsigned lds_off(const void* p) {
  return (unsigned)(unsigned long long)p;
}

// ---------------------------------------------------------------------------
// f32 -> f16 weight conversion (weights are tiny; done once per launch)
// ---------------------------------------------------------------------------
__global__ __launch_bounds__(256) void f32_to_f16_kernel(
    const float* __restrict__ src, _Float16* __restrict__ dst, int n) {
  int i = blockIdx.x * 256 + threadIdx.x;
  if (i < n) dst[i] = (_Float16)src[i];
}

// ---------------------------------------------------------------------------
// Precompute combined attention bias: biasC[wi][head][n(64)][m(64)] =
//   rpb[rel_idx[n,m], head] + attn_mask[wi,n,m]   (n,m < 49)
//   -1e30 for padded columns m >= 49, 0 for padded rows n >= 49.
// ---------------------------------------------------------------------------
__global__ __launch_bounds__(256) void bias_precompute_kernel(
    const int* __restrict__ rel_idx, const float* __restrict__ rpb,
    const float* __restrict__ mask, float* __restrict__ biasC) {
  const int idx = blockIdx.x * 256 + threadIdx.x;  // 64*8*64*64 = 2,097,152
  const int m = idx & 63;
  const int n = (idx >> 6) & 63;
  const int head = (idx >> 12) & 7;
  const int wi = idx >> 15;
  float v;
  if (m >= NTOK)      v = -1e30f;
  else if (n >= NTOK) v = 0.f;
  else v = rpb[rel_idx[n * NTOK + m] * NHEAD + head] +
           mask[wi * (NTOK * NTOK) + n * NTOK + m];
  biasC[idx] = v;
}

// ---------------------------------------------------------------------------
// LN1 + cyclic shift(-3,-3) + window partition: x[B,L,C] f32 -> hwin f16
// (window-ordered rows: row = win*49 + n). One wave per output token.
// ---------------------------------------------------------------------------
__global__ __launch_bounds__(256) void ln1_window_kernel(
    const float* __restrict__ x, const float* __restrict__ g,
    const float* __restrict__ b, _Float16* __restrict__ out) {
  const int wave = threadIdx.x >> 5, lane = threadIdx.x & 31;
  const int t = blockIdx.x * 8 + wave;          // window-ordered row
  const int win = t / NTOK, n = t % NTOK;
  const int bimg = win >> 6, wi = win & 63;
  const int wr = wi >> 3, wc = wi & 7;
  const int ph = wr * 7 + n / 7, pw = wc * 7 + n % 7;   // shifted coords
  const int si = (ph + 3) % 56, sj = (pw + 3) % 56;     // roll(-3): h[p]=x[p+3]
  const float* src = x + ((size_t)bimg * LTOK + si * 56 + sj) * DIMC;

  float vals[8]; float s = 0.f, sq = 0.f;
#pragma unroll
  for (int k = 0; k < 8; ++k) {
    float v = src[lane + k * 32];
    vals[k] = v; s += v; sq += v * v;
  }
#pragma unroll
  for (int off = 1; off < 32; off <<= 1) {
    s  += __shfl_xor(s, off);
    sq += __shfl_xor(sq, off);
  }
  const float mean = s * (1.f / 256.f);
  const float var  = sq * (1.f / 256.f) - mean * mean;
  const float inv  = rsqrtf(var + 1e-5f);
  _Float16* dst = out + (size_t)t * DIMC;
#pragma unroll
  for (int k = 0; k < 8; ++k) {
    int c = lane + k * 32;
    dst[c] = (_Float16)((vals[k] - mean) * inv * g[c] + b[c]);
  }
}

// ---------------------------------------------------------------------------
// LN2 with window-reverse + roll(-3) gather: xproj (window order, f32)
// -> h2 (original token order, f16)
// ---------------------------------------------------------------------------
__global__ __launch_bounds__(256) void ln2_gather_kernel(
    const float* __restrict__ xproj, const float* __restrict__ g,
    const float* __restrict__ b, _Float16* __restrict__ out) {
  const int wave = threadIdx.x >> 5, lane = threadIdx.x & 31;
  const int t = blockIdx.x * 8 + wave;          // original-order row
  const int bimg = t / LTOK; const int r2 = t % LTOK;
  const int i = r2 / 56, j = r2 % 56;
  const int ph = (i + 3) % 56, pw = (j + 3) % 56;  // final[i]=out_h[(i+3)%56]
  const int wr = ph / 7, ir = ph % 7, wc = pw / 7, jc = pw % 7;
  const int win = bimg * NWIN + wr * 8 + wc, n = ir * 7 + jc;
  const float* src = xproj + ((size_t)win * NTOK + n) * DIMC;

  float vals[8]; float s = 0.f, sq = 0.f;
#pragma unroll
  for (int k = 0; k < 8; ++k) {
    float v = src[lane + k * 32];
    vals[k] = v; s += v; sq += v * v;
  }
#pragma unroll
  for (int off = 1; off < 32; off <<= 1) {
    s  += __shfl_xor(s, off);
    sq += __shfl_xor(sq, off);
  }
  const float mean = s * (1.f / 256.f);
  const float var  = sq * (1.f / 256.f) - mean * mean;
  const float inv  = rsqrtf(var + 1e-5f);
  _Float16* dst = out + (size_t)t * DIMC;
#pragma unroll
  for (int k = 0; k < 8; ++k) {
    int c = lane + k * 32;
    dst[c] = (_Float16)((vals[k] - mean) * inv * g[c] + b[c]);
  }
}

// ---------------------------------------------------------------------------
// Generic WMMA GEMM: C[M,N] = A[M,K] (f16) @ W[N,K]^T (f16) + bias, epilogue.
// Block = 4 waves, block tile 64(M)x64(N); wave tile 16x64 -> 4 wmma / k-step.
// EPI: 0 = QKV (scale q cols, f16 out)  1 = plain f32 out
//      2 = exact GELU, f16 out          3 = + residual, f32 out
// ---------------------------------------------------------------------------
template <int EPI>
__global__ __launch_bounds__(128) void wmma_gemm_kernel(
    const _Float16* __restrict__ A, const _Float16* __restrict__ W,
    const float* __restrict__ bias, const float* __restrict__ resid,
    _Float16* __restrict__ outH, float* __restrict__ outF,
    int M, int N, int K) {
  const int lane = threadIdx.x & 31, wave = threadIdx.x >> 5;
  const int l = lane & 15, hi = lane >> 4;
  const int m0 = blockIdx.y * 64 + wave * 16;
  const int n0 = blockIdx.x * 64;

  v8f acc[4] = {};
  const _Float16* arow = A + (size_t)(m0 + l) * K + hi * 8;

  for (int k0 = 0; k0 < K; k0 += 32) {
    Frag af;
    af.h[0] = *(const v8h*)(arow + k0);
    af.h[1] = *(const v8h*)(arow + k0 + 16);
#pragma unroll
    for (int j = 0; j < 4; ++j) {
      const _Float16* brow = W + (size_t)(n0 + j * 16 + l) * K + k0 + hi * 8;
      Frag bf;
      bf.h[0] = *(const v8h*)(brow);
      bf.h[1] = *(const v8h*)(brow + 16);
      acc[j] = __builtin_amdgcn_wmma_f32_16x16x32_f16(
          false, af.v, false, bf.v, (short)0, acc[j], false, false);
    }
  }

#pragma unroll
  for (int j = 0; j < 4; ++j) {
#pragma unroll
    for (int r = 0; r < 8; ++r) {
      const int row = m0 + r + 8 * hi;
      const int col = n0 + j * 16 + l;
      float v = acc[j][r] + bias[col];
      if (EPI == 0) {                       // QKV: q pre-scaled (cols < 256)
        if (col < DIMC) v *= QSCALE;
        outH[(size_t)row * N + col] = (_Float16)v;
      } else if (EPI == 1) {                // proj -> f32
        outF[(size_t)row * N + col] = v;
      } else if (EPI == 2) {                // fc1 + exact GELU -> f16
        v = 0.5f * v * (1.0f + erff(v * 0.70710678118654752f));
        outH[(size_t)row * N + col] = (_Float16)v;
      } else {                              // fc2 + residual -> f32
        v += resid[(size_t)row * N + col];
        outF[(size_t)row * N + col] = v;
      }
    }
  }
}

// ---------------------------------------------------------------------------
// Windowed attention: one block (128 thr = 4 waves) per (window, head).
// q/k/v staged into LDS with GLOBAL_LOAD_ASYNC_TO_LDS_B128 (ASYNCcnt),
// QK^T = 16 wmma, softmax via 16-lane shfl, PV = 16 wmma.
// ---------------------------------------------------------------------------
__global__ __launch_bounds__(128) void attn_kernel(
    const _Float16* __restrict__ qkv, const float* __restrict__ biasC,
    _Float16* __restrict__ xattn) {
  const int head = blockIdx.x;
  const int win  = blockIdx.y;
  const int tid  = threadIdx.x;

  __shared__ __align__(16) _Float16 qS[64 * 40];   // stride 40 halves = 80B
  __shared__ __align__(16) _Float16 kS[64 * 40];
  __shared__ __align__(16) _Float16 vR[64 * 40];   // raw v rows (pre-transpose)
  __shared__ __align__(16) _Float16 vS[32 * 80];   // vS[d][m], stride 160B
  __shared__ __align__(16) _Float16 aS[64 * 80];   // attn f16 staging

  // zero the pad rows (49..63) of q/k/vR so WMMA sees clean zeros
  for (int t = tid; t < 15 * 40; t += 128) {
    qS[49 * 40 + t] = (_Float16)0.f;
    kS[49 * 40 + t] = (_Float16)0.f;
    vR[49 * 40 + t] = (_Float16)0.f;
  }

  // async global->LDS staging: 3 matrices x 49 rows x 4 x b128 chunks
  const _Float16* gb = qkv + (size_t)win * NTOK * 768;  // uniform base (SGPR)
  const unsigned ldsQ = lds_off(qS), ldsK = lds_off(kS), ldsV = lds_off(vR);
  const unsigned hb = (unsigned)(head * HDIM) * 2u;
  for (int t = tid; t < 3 * NTOK * 4; t += 128) {
    const int m  = t / (NTOK * 4);        // 0=q, 1=k, 2=v
    const int rc = t - m * (NTOK * 4);
    const int r = rc >> 2, c = rc & 3;
    const unsigned voff = (unsigned)((r * 768 + m * 256) * 2) + hb +
                          (unsigned)(c * 16);
    const unsigned lbase = (m == 0) ? ldsQ : (m == 1 ? ldsK : ldsV);
    const unsigned loff = lbase + (unsigned)(r * 80 + c * 16);
    asm volatile("global_load_async_to_lds_b128 %0, %1, %2 offset:0"
                 :: "v"(loff), "v"(voff), "s"(gb) : "memory");
  }
  asm volatile("s_wait_asynccnt 0" ::: "memory");
  __syncthreads();

  // transpose v into vS[d][m]
  for (int t = tid; t < 64 * 32; t += 128) {
    const int r = t >> 5, d = t & 31;
    vS[d * 80 + r] = vR[r * 40 + d];
  }
  __syncthreads();

  const int wave = tid >> 5, lane = tid & 31;
  const int l = lane & 15, hi = lane >> 4;
  const int m0 = wave * 16;

  // --- phase 1: attn = q @ k^T (16x64 per wave, K = HD = 32) ---
  v8f acc[4] = {};
  Frag af;
  const _Float16* ap = qS + (m0 + l) * 40 + hi * 8;
  af.h[0] = *(const v8h*)(ap);
  af.h[1] = *(const v8h*)(ap + 16);
#pragma unroll
  for (int j = 0; j < 4; ++j) {
    const _Float16* bp = kS + (j * 16 + l) * 40 + hi * 8;
    Frag bf;
    bf.h[0] = *(const v8h*)(bp);
    bf.h[1] = *(const v8h*)(bp + 16);
    acc[j] = __builtin_amdgcn_wmma_f32_16x16x32_f16(
        false, af.v, false, bf.v, (short)0, acc[j], false, false);
  }

  // precomputed bias + mask (padding baked in) -> unconditional adds
  const float* bC = biasC + (((size_t)(win & 63) * NHEAD + head) << 12);
#pragma unroll
  for (int j = 0; j < 4; ++j) {
    const int m = j * 16 + l;
#pragma unroll
    for (int r = 0; r < 8; ++r) {
      const int n = m0 + r + 8 * hi;
      acc[j][r] += bC[n * 64 + m];
    }
  }

  // --- softmax over the 64 columns (row lives in a 16-lane group x 4 regs) ---
#pragma unroll
  for (int r = 0; r < 8; ++r) {
    float mx = fmaxf(fmaxf(acc[0][r], acc[1][r]), fmaxf(acc[2][r], acc[3][r]));
#pragma unroll
    for (int off = 1; off < 16; off <<= 1) mx = fmaxf(mx, __shfl_xor(mx, off));
    float s = 0.f;
#pragma unroll
    for (int j = 0; j < 4; ++j) {
      float e = __expf(acc[j][r] - mx);
      acc[j][r] = e; s += e;
    }
#pragma unroll
    for (int off = 1; off < 16; off <<= 1) s += __shfl_xor(s, off);
    const float inv = 1.f / s;
#pragma unroll
    for (int j = 0; j < 4; ++j) acc[j][r] *= inv;
  }

  // stage attn as f16 in LDS for the PV matmul's A-fragments
#pragma unroll
  for (int j = 0; j < 4; ++j) {
#pragma unroll
    for (int r = 0; r < 8; ++r) {
      const int n = m0 + r + 8 * hi;
      const int m = j * 16 + l;
      aS[n * 80 + m] = (_Float16)acc[j][r];
    }
  }
  __syncthreads();

  // --- phase 2: out = attn @ v (16x32 per wave, K = 64 in two steps) ---
  v8f oacc[2] = {};
#pragma unroll
  for (int ks = 0; ks < 2; ++ks) {
    const int k0 = ks * 32;
    const _Float16* ap2 = aS + (m0 + l) * 80 + k0 + hi * 8;
    Frag a2;
    a2.h[0] = *(const v8h*)(ap2);
    a2.h[1] = *(const v8h*)(ap2 + 16);
#pragma unroll
    for (int j = 0; j < 2; ++j) {
      const _Float16* bp2 = vS + (j * 16 + l) * 80 + k0 + hi * 8;
      Frag b2;
      b2.h[0] = *(const v8h*)(bp2);
      b2.h[1] = *(const v8h*)(bp2 + 16);
      oacc[j] = __builtin_amdgcn_wmma_f32_16x16x32_f16(
          false, a2.v, false, b2.v, (short)0, oacc[j], false, false);
    }
  }

#pragma unroll
  for (int j = 0; j < 2; ++j) {
#pragma unroll
    for (int r = 0; r < 8; ++r) {
      const int n = m0 + r + 8 * hi;
      if (n < NTOK) {
        const int d = j * 16 + l;
        xattn[((size_t)win * NTOK + n) * DIMC + head * HDIM + d] =
            (_Float16)oacc[j][r];
      }
    }
  }
}

// ---------------------------------------------------------------------------
extern "C" void kernel_launch(void* const* d_in, const int* in_sizes, int n_in,
                              void* d_out, int out_size, void* d_ws, size_t ws_size,
                              hipStream_t stream) {
  (void)in_sizes; (void)n_in; (void)out_size; (void)ws_size;

  const float* x         = (const float*)d_in[0];
  const float* norm1_g   = (const float*)d_in[1];
  const float* norm1_b   = (const float*)d_in[2];
  const float* qkv_w     = (const float*)d_in[3];
  const float* qkv_b     = (const float*)d_in[4];
  const float* rpb_table = (const float*)d_in[5];
  const float* proj_w    = (const float*)d_in[6];
  const float* proj_b    = (const float*)d_in[7];
  const float* norm2_g   = (const float*)d_in[8];
  const float* norm2_b   = (const float*)d_in[9];
  const float* fc1_w     = (const float*)d_in[10];
  const float* fc1_b     = (const float*)d_in[11];
  const float* fc2_w     = (const float*)d_in[12];
  const float* fc2_b     = (const float*)d_in[13];
  const int*   rel_idx   = (const int*)d_in[14];
  const float* attn_mask = (const float*)d_in[15];

  // workspace layout (bytes), with aliasing:
  //   hwin  f16 [100352,256]   (reused as h2 after QKV GEMM consumes it)
  //   qkv   f16 [100352,768]   (qkv..xattn region reused as mlp1 [100352,1024])
  //   xattn f16 [100352,256]
  //   xproj f32 [100352,256]
  //   f16 weight copies + combined bias table
  char* ws = (char*)d_ws;
  const size_t o_hwin  = 0;
  const size_t o_qkv   = o_hwin + (size_t)MROWS * DIMC * 2;
  const size_t o_xattn = o_qkv  + (size_t)MROWS * 768 * 2;
  const size_t o_xproj = o_xattn + (size_t)MROWS * DIMC * 2;
  const size_t o_wts   = o_xproj + (size_t)MROWS * DIMC * 4;
  const size_t o_bias  = o_wts + (size_t)(768 * 256 + 256 * 256 + 1024 * 256 +
                                          256 * 1024) * 2;

  _Float16* hwin  = (_Float16*)(ws + o_hwin);
  _Float16* h2    = hwin;                         // alias
  _Float16* qkvB  = (_Float16*)(ws + o_qkv);
  _Float16* mlp1  = qkvB;                         // alias (qkv+xattn span)
  _Float16* xattn = (_Float16*)(ws + o_xattn);
  float*    xproj = (float*)(ws + o_xproj);
  _Float16* wqkv  = (_Float16*)(ws + o_wts);
  _Float16* wproj = wqkv  + 768 * 256;
  _Float16* wfc1  = wproj + 256 * 256;
  _Float16* wfc2  = wfc1  + 1024 * 256;
  float*    biasC = (float*)(ws + o_bias);        // [64][8][64][64] f32

  // 0) weight conversion + combined bias table
  f32_to_f16_kernel<<<(768 * 256 + 255) / 256, 256, 0, stream>>>(qkv_w, wqkv, 768 * 256);
  f32_to_f16_kernel<<<(256 * 256 + 255) / 256, 256, 0, stream>>>(proj_w, wproj, 256 * 256);
  f32_to_f16_kernel<<<(1024 * 256 + 255) / 256, 256, 0, stream>>>(fc1_w, wfc1, 1024 * 256);
  f32_to_f16_kernel<<<(256 * 1024 + 255) / 256, 256, 0, stream>>>(fc2_w, wfc2, 256 * 1024);
  bias_precompute_kernel<<<(NWIN * NHEAD * 64 * 64) / 256, 256, 0, stream>>>(
      rel_idx, rpb_table, attn_mask, biasC);

  // 1) LN1 + roll + window partition -> hwin f16
  ln1_window_kernel<<<MROWS / 8, 256, 0, stream>>>(x, norm1_g, norm1_b, hwin);

  // 2) QKV GEMM (q pre-scaled): [100352,256] x [768,256]^T -> qkv f16
  wmma_gemm_kernel<0><<<dim3(768 / 64, MROWS / 64), 128, 0, stream>>>(
      hwin, wqkv, qkv_b, nullptr, qkvB, nullptr, MROWS, 768, 256);

  // 3) windowed attention -> xattn f16 [100352,256]
  attn_kernel<<<dim3(NHEAD, BWIN), 128, 0, stream>>>(qkvB, biasC, xattn);

  // 4) proj GEMM: [100352,256] x [256,256]^T -> xproj f32 (window order)
  wmma_gemm_kernel<1><<<dim3(256 / 64, MROWS / 64), 128, 0, stream>>>(
      xattn, wproj, proj_b, nullptr, nullptr, xproj, MROWS, 256, 256);

  // 5) LN2 with window-reverse + roll gather -> h2 f16 (original order)
  ln2_gather_kernel<<<MROWS / 8, 256, 0, stream>>>(xproj, norm2_g, norm2_b, h2);

  // 6) fc1 GEMM + exact GELU: [100352,256] x [1024,256]^T -> mlp1 f16
  wmma_gemm_kernel<2><<<dim3(1024 / 64, MROWS / 64), 128, 0, stream>>>(
      h2, wfc1, fc1_b, nullptr, mlp1, nullptr, MROWS, 1024, 256);

  // 7) fc2 GEMM + residual (shortcut = original x): -> d_out f32
  wmma_gemm_kernel<3><<<dim3(256 / 64, MROWS / 64), 128, 0, stream>>>(
      mlp1, wfc2, fc2_b, x, nullptr, (float*)d_out, MROWS, 256, 1024);
}
